// Informer_82617990906350
// MI455X (gfx1250) — matrix operand order repaired and checked by reading, not compiled, not run on hardware
//
#include <hip/hip_runtime.h>
#include <hip/hip_bf16.h>
#include <math.h>

// ---------------------------------------------------------------------------
// Informer forward for MI455X (gfx1250, wave32).
// All dense projections / FFN GEMMs go through V_WMMA_F32_16X16X32_F16.
// ProbSparse attention (40/35 active queries per head) is VALU - it is <1%
// of total FLOPs.  fp32 master activations, f16 GEMM operands, f32 accum.
// ---------------------------------------------------------------------------

typedef __attribute__((ext_vector_type(16))) _Float16 v16h;
typedef __attribute__((ext_vector_type(8)))  _Float16 v8h;
typedef __attribute__((ext_vector_type(8)))  float    v8f;

#define D_MODEL 512
#define DFF     2048
#define NHEADS  8
#define EDIM    64
#define BATCH   2
#define L_ENC   2048
#define L_DEC   1024
#define ENC_M   (BATCH * L_ENC)   // 4096
#define DEC_M   (BATCH * L_DEC)   // 2048

static inline int cdiv(long a, long b) { return (int)((a + b - 1) / b); }

// ---------------------------------------------------------------------------
// WMMA GEMM: C[M,N] = A[M,K](f16) * Bt[N,K](f16)^T + bias (+ residual)(+GELU)
// grid = (N/64, M/64), block = 128 (4 waves). Wave w -> rows [by*64+w*16,+16),
// cols [bx*64, +64) as 4 accumulators.
// ---------------------------------------------------------------------------
__global__ __launch_bounds__(128)
void k_gemm_wmma(const _Float16* __restrict__ A, const _Float16* __restrict__ Bt,
                 const float* __restrict__ bias, const float* __restrict__ resid,
                 float* __restrict__ C, int M, int N, int K, int doGelu)
{
    const int lane = threadIdx.x & 31;
    const int wave = threadIdx.x >> 5;
    const int row0 = blockIdx.y * 64 + wave * 16;
    const int col0 = blockIdx.x * 64;
    const int m16  = lane & 15;
    const int kh   = lane >> 4;          // half-wave id

    const long arowOff = (long)(row0 + m16) * K;

    v8f acc[4] = {};
    for (int k0 = 0; k0 < K; k0 += 32) {
        // A fragment (16x32 f16): this lane holds K = kh*16..+15
        const _Float16* ap = A + arowOff + k0 + kh * 16;
        __builtin_prefetch(ap + 64, 0, 1);            // global_prefetch_b8
        v8h alo = *(const v8h*)(ap);
        v8h ahi = *(const v8h*)(ap + 8);
        v16h a;
#pragma unroll
        for (int i = 0; i < 8; ++i) { a[i] = alo[i]; a[8 + i] = ahi[i]; }

#pragma unroll
        for (int t = 0; t < 4; ++t) {
            // B fragment (32x16): lanes 0-15 K=0..15, lanes 16-31 K=16..31,
            // N = lane&15.  Bt is [N,K] so the 16 halves are contiguous.
            const _Float16* bp = Bt + (long)(col0 + t * 16 + m16) * K + k0 + kh * 16;
            v16h b = *(const v16h*)bp;
            acc[t] = __builtin_amdgcn_wmma_f32_16x16x32_f16(
                false, a, false, b, (short)0, acc[t], false, false);
        }
    }
#pragma unroll
    for (int t = 0; t < 4; ++t) {
        const int col = col0 + t * 16 + m16;
        const float bv = bias ? bias[col] : 0.0f;
#pragma unroll
        for (int r = 0; r < 8; ++r) {
            const int row = row0 + kh * 8 + r;   // C layout: VGPR r -> M=kh*8+r
            float x = acc[t][r] + bv;
            if (doGelu) x = 0.5f * x * (1.0f + erff(x * 0.70710678118f));
            if (resid) x += resid[(long)row * N + col];
            C[(long)row * N + col] = x;
        }
    }
}

// --------------------------- elementwise helpers ---------------------------

__global__ void k_cvt_f16(const float* __restrict__ X, _Float16* __restrict__ Y, long n)
{
    long i = (long)blockIdx.x * blockDim.x + threadIdx.x;
    if (i < n) Y[i] = (_Float16)X[i];
}

// W[K,N] fp32 -> Wt[N,K] f16 (transposed for the B operand)
__global__ void k_wt_f16(const float* __restrict__ W, _Float16* __restrict__ Wt, int K, int N)
{
    long i = (long)blockIdx.x * blockDim.x + threadIdx.x;
    if (i >= (long)K * N) return;
    int n = (int)(i % N), k = (int)(i / N);
    Wt[(long)n * K + k] = (_Float16)W[i];
}

// token_embed (circular width-3 conv on the single input feature) + pos embed
__global__ void k_embed(const float* __restrict__ xin, const float* __restrict__ W3,
                        float* __restrict__ out, int B_, int L)
{
    long i = (long)blockIdx.x * blockDim.x + threadIdx.x;
    if (i >= (long)B_ * L * D_MODEL) return;
    int c = (int)(i % D_MODEL);
    long r = i / D_MODEL;
    int l = (int)(r % L), b = (int)(r / L);
    int lm = (l == 0) ? L - 1 : l - 1;
    int lp = (l == L - 1) ? 0 : l + 1;
    float x0 = xin[(long)lm * B_ + b]; if (!(x0 == x0)) x0 = 0.0f;
    float x1 = xin[(long)l  * B_ + b]; if (!(x1 == x1)) x1 = 0.0f;
    float x2 = xin[(long)lp * B_ + b]; if (!(x2 == x2)) x2 = 0.0f;
    float v = x0 * W3[c * 3 + 0] + x1 * W3[c * 3 + 1] + x2 * W3[c * 3 + 2];
    float freq = __expf(-(float)(c & ~1) * (9.210340372f / (float)D_MODEL));
    float ang  = (float)l * freq;
    v += (c & 1) ? __cosf(ang) : __sinf(ang);
    out[i] = v;
}

// LayerNorm over 512, one row per block; in-place safe (row cached in LDS).
__global__ __launch_bounds__(256)
void k_layernorm(const float* __restrict__ X, float* __restrict__ Y,
                 const float* __restrict__ g, const float* __restrict__ b, int rows)
{
    __shared__ float row[D_MODEL];
    __shared__ float red[256];
    int r = blockIdx.x; if (r >= rows) return;
    int t = threadIdx.x;
    float v0 = X[(long)r * D_MODEL + t];
    float v1 = X[(long)r * D_MODEL + 256 + t];
    row[t] = v0; row[256 + t] = v1;
    red[t] = v0 + v1; __syncthreads();
    for (int k = 128; k > 0; k >>= 1) { if (t < k) red[t] += red[t + k]; __syncthreads(); }
    float mean = red[0] * (1.0f / D_MODEL); __syncthreads();
    float d0 = v0 - mean, d1 = v1 - mean;
    red[t] = d0 * d0 + d1 * d1; __syncthreads();
    for (int k = 128; k > 0; k >>= 1) { if (t < k) red[t] += red[t + k]; __syncthreads(); }
    float rs = rsqrtf(red[0] * (1.0f / D_MODEL) + 1e-5f);
    Y[(long)r * D_MODEL + t]       = (row[t]       - mean) * rs * g[t]       + b[t];
    Y[(long)r * D_MODEL + 256 + t] = (row[256 + t] - mean) * rs * g[256 + t] + b[256 + t];
}

// ------------------------ ProbSparse attention -----------------------------

// M[bh,lq] = max_u(q.k_sample) - sum_u(q.k_sample)/Lk   (one wave per query)
__global__ __launch_bounds__(256)
void k_attn_m(const float* __restrict__ Q, const float* __restrict__ Kt,
              float* __restrict__ Mbuf, int B_, int Lq, int Lk, int U)
{
    int gw = blockIdx.x * 8 + (threadIdx.x >> 5);
    int lane = threadIdx.x & 31;
    if (gw >= B_ * NHEADS * Lq) return;
    int lq = gw % Lq, bh = gw / Lq, h = bh % NHEADS, b = bh / NHEADS;
    const float* qr = Q + ((long)(b * Lq + lq)) * D_MODEL + h * EDIM;
    float mx = -INFINITY, sm = 0.0f;
    for (int u = lane; u < U; u += 32) {
        unsigned idx = (unsigned)lq * 2654435761u + (unsigned)u * 40503u + 0x9E3779B9u;
        idx = (idx ^ (idx >> 16)) % (unsigned)Lk;
        const float* kr = Kt + ((long)(b * Lk + (int)idx)) * D_MODEL + h * EDIM;
        float d = 0.0f;
        for (int e = 0; e < EDIM; ++e) d += qr[e] * kr[e];
        mx = fmaxf(mx, d); sm += d;
    }
    for (int m = 16; m; m >>= 1) {
        mx = fmaxf(mx, __shfl_xor(mx, m, 32));
        sm += __shfl_xor(sm, m, 32);
    }
    if (lane == 0) Mbuf[(long)bh * Lq + lq] = mx - sm / (float)Lk;
}

// top-u indices per (b,h) by repeated argmax (u<=40, Lq<=2048: trivial work)
__global__ __launch_bounds__(256)
void k_topk(const float* __restrict__ Mbuf, int* __restrict__ mtop, int Lq, int u)
{
    __shared__ float vals[2048];
    __shared__ float rv[256];
    __shared__ int   ri[256];
    int bh = blockIdx.x, t = threadIdx.x;
    for (int i = t; i < Lq; i += 256) vals[i] = Mbuf[(long)bh * Lq + i];
    __syncthreads();
    for (int j = 0; j < u; ++j) {
        float best = -INFINITY; int bi = 0;
        for (int i = t; i < Lq; i += 256) { float v = vals[i]; if (v > best) { best = v; bi = i; } }
        rv[t] = best; ri[t] = bi; __syncthreads();
        for (int k = 128; k > 0; k >>= 1) {
            if (t < k && rv[t + k] > rv[t]) { rv[t] = rv[t + k]; ri[t] = ri[t + k]; }
            __syncthreads();
        }
        if (t == 0) { mtop[bh * u + j] = ri[0]; vals[ri[0]] = -INFINITY; }
        __syncthreads();
    }
}

// default context: causal cumsum(V)   ctx layout [B,H,L,E]
__global__ void k_ctx_cumsum(const float* __restrict__ V, float* __restrict__ ctx, int B_, int L)
{
    int i = blockIdx.x * blockDim.x + threadIdx.x;
    if (i >= B_ * NHEADS * EDIM) return;
    int e = i & 63, h = (i >> 6) & 7, b = i >> 9;
    float acc = 0.0f;
    for (int l = 0; l < L; ++l) {
        acc += V[((long)(b * L + l)) * D_MODEL + h * EDIM + e];
        ctx[(((long)(b * NHEADS + h) * L) + l) * EDIM + e] = acc;
    }
}

// default context: broadcast mean(V) (cross attention)
__global__ void k_ctx_mean(const float* __restrict__ V, float* __restrict__ ctx,
                           int B_, int Lk, int Lq)
{
    int i = blockIdx.x * blockDim.x + threadIdx.x;
    if (i >= B_ * NHEADS * EDIM) return;
    int e = i & 63, h = (i >> 6) & 7, b = i >> 9;
    float s = 0.0f;
    for (int l = 0; l < Lk; ++l) s += V[((long)(b * Lk + l)) * D_MODEL + h * EDIM + e];
    s /= (float)Lk;
    for (int l = 0; l < Lq; ++l)
        ctx[(((long)(b * NHEADS + h) * Lq) + l) * EDIM + e] = s;
}

// scores+softmax+update for one selected query; scatter into ctx
__global__ __launch_bounds__(256)
void k_attn_upd(const float* __restrict__ Q, const float* __restrict__ Kt,
                const float* __restrict__ V, const int* __restrict__ mtop,
                float* __restrict__ ctx, int Lq, int Lk, int u, int causal)
{
    __shared__ float sc[2048];
    __shared__ float qs[EDIM];
    __shared__ float red[256];
    __shared__ float part[4][EDIM];
    int bh = blockIdx.x / u, j = blockIdx.x % u;
    int h = bh & 7, b = bh >> 3;
    int t = threadIdx.x;
    int qrow = mtop[bh * u + j];
    if (t < EDIM) qs[t] = Q[((long)(b * Lq + qrow)) * D_MODEL + h * EDIM + t];
    __syncthreads();
    int limit = causal ? (qrow + 1) : Lk;
    float mx = -INFINITY;
    for (int c = t; c < limit; c += 256) {
        const float* kr = Kt + ((long)(b * Lk + c)) * D_MODEL + h * EDIM;
        float d = 0.0f;
        for (int e = 0; e < EDIM; ++e) d += qs[e] * kr[e];
        d *= 0.125f;                    // 1/sqrt(64)
        sc[c] = d; mx = fmaxf(mx, d);
    }
    red[t] = mx; __syncthreads();
    for (int k = 128; k > 0; k >>= 1) { if (t < k) red[t] = fmaxf(red[t], red[t + k]); __syncthreads(); }
    mx = red[0]; __syncthreads();
    float se = 0.0f;
    for (int c = t; c < limit; c += 256) { float ex = __expf(sc[c] - mx); sc[c] = ex; se += ex; }
    red[t] = se; __syncthreads();
    for (int k = 128; k > 0; k >>= 1) { if (t < k) red[t] += red[t + k]; __syncthreads(); }
    se = red[0]; __syncthreads();
    int e = t & 63, g = t >> 6;
    float acc = 0.0f;
    for (int c = g; c < limit; c += 4)
        acc += sc[c] * V[((long)(b * Lk + c)) * D_MODEL + h * EDIM + e];
    part[g][e] = acc; __syncthreads();
    if (g == 0) {
        float s = part[0][e] + part[1][e] + part[2][e] + part[3][e];
        ctx[(((long)(b * NHEADS + h) * Lq) + qrow) * EDIM + e] = s / se;
    }
}

// ctx [B,H,L,E] -> f16 O-proj input. mix=1: flat copy ((B,H,L,E)->(B,L,H*E)
// reinterpretation as in the reference); mix=0: per-token head gather.
__global__ void k_ctx_to_f16(const float* __restrict__ ctx, _Float16* __restrict__ out,
                             int B_, int L, int mix)
{
    long n = (long)B_ * L * D_MODEL;
    long i = (long)blockIdx.x * blockDim.x + threadIdx.x;
    if (i >= n) return;
    if (mix) { out[i] = (_Float16)ctx[i]; return; }
    long per = (long)L * D_MODEL;
    int b = (int)(i / per); long r = i % per;
    int l = (int)(r / D_MODEL), c = (int)(r % D_MODEL);
    int h = c >> 6, e = c & 63;
    out[i] = (_Float16)ctx[((long)(b * NHEADS + h) * L + l) * EDIM + e];
}

// final projection d->1, output transposed to (L,B,1)
__global__ __launch_bounds__(256)
void k_proj(const float* __restrict__ X, const float* __restrict__ w,
            const float* __restrict__ bb, float* __restrict__ out, int B_, int L)
{
    int gw = blockIdx.x * 8 + (threadIdx.x >> 5);
    int lane = threadIdx.x & 31;
    if (gw >= B_ * L) return;
    int l = gw % L, b = gw / L;
    const float* xr = X + (long)gw * D_MODEL;
    float s = 0.0f;
    for (int j = lane; j < D_MODEL; j += 32) s += xr[j] * w[j];
    for (int m = 16; m; m >>= 1) s += __shfl_xor(s, m, 32);
    if (lane == 0) out[(long)l * B_ + b] = s + bb[0];
}

// ---------------------------------------------------------------------------
// Host orchestration.
// Param leaves follow JAX pytree order (sorted dict keys) inside `params`:
//   2: dec_conv
//   3..28 / 29..54: dec_layers[0/1]
//       +0..7 cross{k.b,k.w,o.b,o.w,q.b,q.w,v.b,v.w} +8,9 ff1{b,w}
//       +10,11 ff2{b,w} +12..17 ln1/2/3{b,g} +18..25 self{k,o,q,v}{b,w}
//   55,56: dec_norm{b,g}   57: enc_conv
//   58..73 / 74..89: enc_layers[0/1]
//       +0..7 attn{k,o,q,v}{b,w} +8..11 ff1/ff2{b,w} +12..15 ln1/ln2{b,g}
//   90,91: enc_norm{b,g}   92,93: proj{b,w}
// ---------------------------------------------------------------------------
extern "C" void kernel_launch(void* const* d_in, const int* in_sizes, int n_in,
                              void* d_out, int out_size, void* d_ws, size_t ws_size,
                              hipStream_t stream)
{
    (void)in_sizes; (void)n_in; (void)out_size; (void)ws_size;
    const float* xe = (const float*)d_in[0];
    const float* xd = (const float*)d_in[1];
    auto P = [&](int i) { return (const float*)d_in[i]; };

    // workspace carve-out
    size_t off = 0;
    auto take = [&](size_t bytes) -> char* {
        size_t r = (off + 255) & ~(size_t)255; off = r + bytes; return (char*)d_ws + r;
    };
    float*    enc_x = (float*)   take((size_t)ENC_M * D_MODEL * 4);
    float*    dec_x = (float*)   take((size_t)DEC_M * D_MODEL * 4);
    float*    tmp1  = (float*)   take((size_t)ENC_M * D_MODEL * 4);
    float*    qbuf  = (float*)   take((size_t)ENC_M * D_MODEL * 4);
    float*    kbuf  = (float*)   take((size_t)ENC_M * D_MODEL * 4);
    float*    vbuf  = (float*)   take((size_t)ENC_M * D_MODEL * 4);
    float*    ctx   = (float*)   take((size_t)ENC_M * D_MODEL * 4);
    float*    hid   = (float*)   take((size_t)ENC_M * DFF * 4);
    _Float16* a16   = (_Float16*)take((size_t)ENC_M * DFF * 2);
    _Float16* w16   = (_Float16*)take((size_t)DFF * D_MODEL * 2);
    float*    mbuf  = (float*)   take((size_t)BATCH * NHEADS * L_ENC * 4);
    int*      mtop  = (int*)     take((size_t)BATCH * NHEADS * 40 * 4);

    auto cvt = [&](const float* s, _Float16* d, long n) {
        k_cvt_f16<<<cdiv(n, 256), 256, 0, stream>>>(s, d, n);
    };
    auto gemm = [&](const _Float16* A, int M, int K, int N, int wIdx, int bIdx,
                    const float* resid, float* Cout, int gelu) {
        k_wt_f16<<<cdiv((long)K * N, 256), 256, 0, stream>>>(P(wIdx), w16, K, N);
        dim3 g(N / 64, M / 64);
        k_gemm_wmma<<<g, 128, 0, stream>>>(A, w16, P(bIdx), resid, Cout, M, N, K, gelu);
    };
    auto ln = [&](const float* s, float* d, int gI, int bI, int rows) {
        k_layernorm<<<rows, 256, 0, stream>>>(s, d, P(gI), P(bI), rows);
    };

    // ---------------- encoder ----------------
    k_embed<<<cdiv((long)ENC_M * D_MODEL, 256), 256, 0, stream>>>(xe, P(57), enc_x, BATCH, L_ENC);

    for (int li = 0; li < 2; ++li) {
        int o = 58 + 16 * li;
        cvt(enc_x, a16, (long)ENC_M * D_MODEL);
        gemm(a16, ENC_M, D_MODEL, D_MODEL, o + 5, o + 4, nullptr, qbuf, 0);  // q
        gemm(a16, ENC_M, D_MODEL, D_MODEL, o + 1, o + 0, nullptr, kbuf, 0);  // k
        gemm(a16, ENC_M, D_MODEL, D_MODEL, o + 7, o + 6, nullptr, vbuf, 0);  // v
        k_attn_m<<<cdiv(BATCH * NHEADS * L_ENC, 8), 256, 0, stream>>>(qbuf, kbuf, mbuf, BATCH, L_ENC, L_ENC, 40);
        k_topk<<<BATCH * NHEADS, 256, 0, stream>>>(mbuf, mtop, L_ENC, 40);
        k_ctx_cumsum<<<cdiv(BATCH * NHEADS * EDIM, 256), 256, 0, stream>>>(vbuf, ctx, BATCH, L_ENC);
        k_attn_upd<<<BATCH * NHEADS * 40, 256, 0, stream>>>(qbuf, kbuf, vbuf, mtop, ctx, L_ENC, L_ENC, 40, 1);
        k_ctx_to_f16<<<cdiv((long)ENC_M * D_MODEL, 256), 256, 0, stream>>>(ctx, a16, BATCH, L_ENC, 0);
        gemm(a16, ENC_M, D_MODEL, D_MODEL, o + 3, o + 2, enc_x, tmp1, 0);    // o + residual
        ln(tmp1, enc_x, o + 13, o + 12, ENC_M);                              // ln1
        cvt(enc_x, a16, (long)ENC_M * D_MODEL);
        gemm(a16, ENC_M, D_MODEL, DFF, o + 9, o + 8, nullptr, hid, 1);       // ff1 + gelu
        cvt(hid, a16, (long)ENC_M * DFF);
        gemm(a16, ENC_M, DFF, D_MODEL, o + 11, o + 10, enc_x, tmp1, 0);      // ff2 + residual
        ln(tmp1, enc_x, o + 15, o + 14, ENC_M);                              // ln2
    }
    ln(enc_x, enc_x, 91, 90, ENC_M);                                         // enc_norm

    // ---------------- decoder ----------------
    k_embed<<<cdiv((long)DEC_M * D_MODEL, 256), 256, 0, stream>>>(xd, P(2), dec_x, BATCH, L_DEC);

    for (int li = 0; li < 2; ++li) {
        int o = 3 + 26 * li;
        // self attention (masked, mix=True), u = U = 35
        cvt(dec_x, a16, (long)DEC_M * D_MODEL);
        gemm(a16, DEC_M, D_MODEL, D_MODEL, o + 23, o + 22, nullptr, qbuf, 0);
        gemm(a16, DEC_M, D_MODEL, D_MODEL, o + 19, o + 18, nullptr, kbuf, 0);
        gemm(a16, DEC_M, D_MODEL, D_MODEL, o + 25, o + 24, nullptr, vbuf, 0);
        k_attn_m<<<cdiv(BATCH * NHEADS * L_DEC, 8), 256, 0, stream>>>(qbuf, kbuf, mbuf, BATCH, L_DEC, L_DEC, 35);
        k_topk<<<BATCH * NHEADS, 256, 0, stream>>>(mbuf, mtop, L_DEC, 35);
        k_ctx_cumsum<<<cdiv(BATCH * NHEADS * EDIM, 256), 256, 0, stream>>>(vbuf, ctx, BATCH, L_DEC);
        k_attn_upd<<<BATCH * NHEADS * 35, 256, 0, stream>>>(qbuf, kbuf, vbuf, mtop, ctx, L_DEC, L_DEC, 35, 1);
        k_ctx_to_f16<<<cdiv((long)DEC_M * D_MODEL, 256), 256, 0, stream>>>(ctx, a16, BATCH, L_DEC, 1);
        gemm(a16, DEC_M, D_MODEL, D_MODEL, o + 21, o + 20, dec_x, tmp1, 0);
        ln(tmp1, dec_x, o + 13, o + 12, DEC_M);                              // ln1
        // cross attention (unmasked, mix=False), u=35 (Lq=1024), U=40 (Lk=2048)
        cvt(dec_x, a16, (long)DEC_M * D_MODEL);
        gemm(a16, DEC_M, D_MODEL, D_MODEL, o + 5, o + 4, nullptr, qbuf, 0);
        cvt(enc_x, a16, (long)ENC_M * D_MODEL);
        gemm(a16, ENC_M, D_MODEL, D_MODEL, o + 1, o + 0, nullptr, kbuf, 0);
        gemm(a16, ENC_M, D_MODEL, D_MODEL, o + 7, o + 6, nullptr, vbuf, 0);
        k_attn_m<<<cdiv(BATCH * NHEADS * L_DEC, 8), 256, 0, stream>>>(qbuf, kbuf, mbuf, BATCH, L_DEC, L_ENC, 40);
        k_topk<<<BATCH * NHEADS, 256, 0, stream>>>(mbuf, mtop, L_DEC, 35);
        k_ctx_mean<<<cdiv(BATCH * NHEADS * EDIM, 256), 256, 0, stream>>>(vbuf, ctx, BATCH, L_ENC, L_DEC);
        k_attn_upd<<<BATCH * NHEADS * 35, 256, 0, stream>>>(qbuf, kbuf, vbuf, mtop, ctx, L_DEC, L_ENC, 35, 0);
        k_ctx_to_f16<<<cdiv((long)DEC_M * D_MODEL, 256), 256, 0, stream>>>(ctx, a16, BATCH, L_DEC, 0);
        gemm(a16, DEC_M, D_MODEL, D_MODEL, o + 3, o + 2, dec_x, tmp1, 0);
        ln(tmp1, dec_x, o + 15, o + 14, DEC_M);                              // ln2
        // FFN
        cvt(dec_x, a16, (long)DEC_M * D_MODEL);
        gemm(a16, DEC_M, D_MODEL, DFF, o + 9, o + 8, nullptr, hid, 1);
        cvt(hid, a16, (long)DEC_M * DFF);
        gemm(a16, DEC_M, DFF, D_MODEL, o + 11, o + 10, dec_x, tmp1, 0);
        ln(tmp1, dec_x, o + 17, o + 16, DEC_M);                              // ln3
    }
    ln(dec_x, dec_x, 56, 55, DEC_M);                                         // dec_norm

    k_proj<<<cdiv(BATCH * L_DEC, 8), 256, 0, stream>>>(dec_x, P(93), P(92), (float*)d_out, BATCH, L_DEC);
}
